// SimpleHeadAttentionMultiLatent_21028159881346
// MI455X (gfx1250) — compile-verified, matrix operand order checked by reading, not run on hardware
//
#include <hip/hip_runtime.h>

// ---------------- problem constants ----------------
constexpr int  Bsz = 4, Sq = 2048, Dm = 512, Hh = 8, HDm = 64, Lm = 128;
constexpr long OUT_OFF = 0;
constexpr long AW_OFF  = (long)Bsz * Sq * Dm;                    // out
constexpr long QK_OFF  = AW_OFF + (long)Bsz * Hh * Sq * Sq;      // attn_weights
constexpr long Q_OFF   = QK_OFF + (long)Bsz * Hh * Sq * Sq;      // qk_values
constexpr long K_OFF   = Q_OFF  + (long)Bsz * Hh * Sq * Lm;      // q
constexpr long V_OFF   = K_OFF  + (long)Bsz * Sq * Lm;           // k
constexpr float SCALE  = 0.08838834764831845f;                   // 1/sqrt(128)

// ---------------- types ----------------
typedef __attribute__((ext_vector_type(16))) __bf16       v16bf;
typedef __attribute__((ext_vector_type(8)))  float        v8f;
typedef __attribute__((ext_vector_type(4)))  unsigned int u32x4;

union FragBf { v16bf v; u32x4 q[2]; };

__device__ __forceinline__ unsigned short f2bf(float f) {
  unsigned u = __builtin_bit_cast(unsigned, f);
  u += 0x7FFFu + ((u >> 16) & 1u);           // round-to-nearest-even
  return (unsigned short)(u >> 16);
}

// LDS byte offset of a __shared__ object (flat LDS pointer low 32 bits)
__device__ __forceinline__ unsigned lds_off(const void* p) {
  return (unsigned)(size_t)p;
}
// CDNA5 async global->LDS copy, 16B per lane, tracked by ASYNCcnt
__device__ __forceinline__ void async_copy_b128(unsigned lds, const void* gaddr) {
  asm volatile("global_load_async_to_lds_b128 %0, %1, off"
               :: "v"(lds), "v"(gaddr) : "memory");
}

// ---------------- f32 -> bf16 convert ----------------
__global__ void conv_bf16_kernel(const float* __restrict__ in,
                                 unsigned short* __restrict__ out, long n) {
  long i = (long)blockIdx.x * 256 + threadIdx.x;
  if (i < n) out[i] = f2bf(in[i]);
}

// ---------------- f32 (KxN) -> bf16 (NxK) transpose-convert ----------------
__global__ void convT_bf16_kernel(const float* __restrict__ in,
                                  unsigned short* __restrict__ out, int K, int N) {
  long i = (long)blockIdx.x * 256 + threadIdx.x;
  if (i >= (long)K * N) return;
  int k = (int)(i / N), n = (int)(i % N);
  out[(long)n * K + k] = f2bf(in[i]);
}

// ---------------- v (BH,S,HD) -> vT (BH,HD,S) bf16 ----------------
__global__ void transpose_v_kernel(const unsigned short* __restrict__ v,
                                   unsigned short* __restrict__ vT) {
  long total = (long)Bsz * Hh * Sq * HDm;
  long i = (long)blockIdx.x * 256 + threadIdx.x;
  if (i >= total) return;
  int d  = (int)(i % HDm);
  long t = i / HDm;
  int s  = (int)(t % Sq);
  int z  = (int)(t / Sq);
  vT[((long)z * HDm + d) * Sq + s] = v[i];
}

// ---------------- generic NT bf16 WMMA GEMM, async double-buffered ----------
// C(M,N) = alpha * A(MxK) @ B(NxK)^T + bias.  64x64 tile / 8 waves; each wave
// 16x32 via two f32 accumulators.  LDS staging via global_load_async_to_lds,
// double buffered (s_wait_asynccnt pipelining).
__global__ __launch_bounds__(256) void gemm_bf16_wmma(
    const unsigned short* __restrict__ A, int lda,
    const unsigned short* __restrict__ Bm, int ldb,
    float* outF, unsigned short* outB,
    const float* __restrict__ bias, float alpha,
    int M, int N, int K, int cmode, int cS, int cHD, int cH) {
  __shared__ alignas(16) unsigned short As[2][64 * 32];
  __shared__ alignas(16) unsigned short Bs[2][64 * 32]; // stored [n][k]
  int bm = blockIdx.y * 64, bn = blockIdx.x * 64;
  int tid = threadIdx.x, wave = tid >> 5, lane = tid & 31;
  int hi = lane >> 4, lr = lane & 15;
  int wm = (wave >> 1) * 16, wn = (wave & 1) * 32;

  // per-thread staging slots: 64 rows x 32 cols, 16B per thread
  int r = tid >> 2, c = (tid & 3) * 8;
  const unsigned short* agp = A + (long)(bm + r) * lda + c;
  const unsigned short* bgp = Bm + (long)(bn + r) * ldb + c;
  unsigned aoff[2] = { lds_off(&As[0][r * 32 + c]), lds_off(&As[1][r * 32 + c]) };
  unsigned boff[2] = { lds_off(&Bs[0][r * 32 + c]), lds_off(&Bs[1][r * 32 + c]) };

  v8f acc0 = {}, acc1 = {};
  int nk = K / 32;
  // prologue: stage tile 0 into buffer 0
  async_copy_b128(aoff[0], agp);
  async_copy_b128(boff[0], bgp);

  for (int kt = 0; kt < nk; ++kt) {
    int buf = kt & 1;
    if (kt + 1 < nk) {  // issue next tile into other buffer, then retire current
      async_copy_b128(aoff[buf ^ 1], agp + (kt + 1) * 32);
      async_copy_b128(boff[buf ^ 1], bgp + (kt + 1) * 32);
      asm volatile("s_wait_asynccnt 0x2" ::: "memory");
    } else {
      asm volatile("s_wait_asynccnt 0x0" ::: "memory");
    }
    __syncthreads();   // tile kt visible to all waves

    FragBf a, b0, b1;
    int ao = (wm + lr) * 32 + hi * 8;
    a.q[0] = *(const u32x4*)&As[buf][ao];
    a.q[1] = *(const u32x4*)&As[buf][ao + 16];
    int bo0 = (wn + lr) * 32 + hi * 8;
    b0.q[0] = *(const u32x4*)&Bs[buf][bo0];
    b0.q[1] = *(const u32x4*)&Bs[buf][bo0 + 16];
    int bo1 = (wn + 16 + lr) * 32 + hi * 8;
    b1.q[0] = *(const u32x4*)&Bs[buf][bo1];
    b1.q[1] = *(const u32x4*)&Bs[buf][bo1 + 16];
    acc0 = __builtin_amdgcn_wmma_f32_16x16x32_bf16(false, a.v, false, b0.v,
                                                   (short)0, acc0, false, false);
    acc1 = __builtin_amdgcn_wmma_f32_16x16x32_bf16(false, a.v, false, b1.v,
                                                   (short)0, acc1, false, false);
    __syncthreads();   // all reads of buf done before it is overwritten
  }

#pragma unroll
  for (int vg = 0; vg < 8; ++vg) {
    int m = bm + wm + vg + hi * 8;               // C layout: lanes16-31 -> M+8
#pragma unroll
    for (int t = 0; t < 2; ++t) {
      int n = bn + wn + t * 16 + lr;
      float val = (t ? acc1[vg] : acc0[vg]) * alpha + (bias ? bias[n] : 0.0f);
      long idx;
      if (cmode == 0) {
        idx = (long)m * N + n;
      } else {                                    // (B,H,S,hd) head split
        int bb = m / cS, sr = m % cS;
        int hh = n / cHD, ll = n % cHD;
        idx = (((long)(bb * cH + hh) * cS + sr) * cHD) + ll;
      }
      if (outF) __builtin_nontemporal_store(val, &outF[idx]);  // write-only f32
      if (outB) outB[idx] = f2bf(val);                         // re-read bf16
    }
  }
}

// ---------------- fused logits-GEMM + mask + softmax + P@V --------------
// One block per (b,h,16-row query tile).  Logits tile (16x2048 f32) lives in
// LDS; qk_values and attn_weights are streamed out once with NT stores; the
// bf16 probability tile feeds the P@V WMMA directly from LDS.
__global__ __launch_bounds__(256) void mla_attn_fused(
    const unsigned short* __restrict__ qb,   // (B,H,S,L) bf16
    const unsigned short* __restrict__ kb,   // (B,S,L)   bf16
    const unsigned short* __restrict__ vT,   // (B*H,HD,S) bf16
    const float* __restrict__ mask,          // (B,S)
    float* __restrict__ qk_out,              // (B,H,S,S)
    float* __restrict__ attnw,               // (B,H,S,S)
    unsigned short* __restrict__ attnB) {    // (B,S,H*HD) bf16
  extern __shared__ char smem[];
  float*          P  = (float*)smem;                                 // 128KB
  unsigned short* W  = (unsigned short*)(smem + 16 * Sq * 4);        // 64KB
  unsigned short* Qs = (unsigned short*)(smem + 16 * Sq * 4 + 16 * Sq * 2); // 4KB
  float* partial = (float*)((char*)Qs + 16 * Lm * 2);                // 4KB
  float* rowinv  = partial + 1024;                                   // 64B

  int b = blockIdx.z, h = blockIdx.y, row0 = blockIdx.x * 16;
  int tid = threadIdx.x, wave = tid >> 5, lane = tid & 31;
  int hi = lane >> 4, lr = lane & 15;
  const float* mrow = mask + (long)b * Sq;
  long qkbase = ((long)(b * Hh + h) * Sq + row0) * Sq;

  // stage 16x128 q tile to LDS (one b128 per thread)
  {
    int qr = tid >> 4, qc = (tid & 15) * 8;
    *(u32x4*)&Qs[qr * Lm + qc] =
        *(const u32x4*)(qb + (((long)(b * Hh + h) * Sq + row0 + qr) * Lm + qc));
  }
  __syncthreads();

  // ---- phase 1: logits tile 16x2048 = Qs @ k^T (WMMA), to LDS + NT global ----
  {
    FragBf a[4];
#pragma unroll
    for (int kc = 0; kc < 4; ++kc) {
      int ao = lr * Lm + kc * 32 + hi * 8;
      a[kc].q[0] = *(const u32x4*)&Qs[ao];
      a[kc].q[1] = *(const u32x4*)&Qs[ao + 16];
    }
    const unsigned short* kbase = kb + (long)b * Sq * Lm;
    for (int t = 0; t < 16; ++t) {               // 8 waves x 16 N-tiles = 2048
      int n0 = (wave * 16 + t) * 16;
      const unsigned short* krow = kbase + (long)(n0 + lr) * Lm;
      v8f acc = {};
#pragma unroll
      for (int kc = 0; kc < 4; ++kc) {
        FragBf bf;
        bf.q[0] = *(const u32x4*)(krow + kc * 32 + hi * 8);
        bf.q[1] = *(const u32x4*)(krow + kc * 32 + hi * 8 + 16);
        acc = __builtin_amdgcn_wmma_f32_16x16x32_bf16(false, a[kc].v, false, bf.v,
                                                      (short)0, acc, false, false);
      }
#pragma unroll
      for (int vg = 0; vg < 8; ++vg) {
        int m = vg + hi * 8, n = n0 + lr;
        float val = acc[vg] * SCALE;
        P[m * Sq + n] = val;
        __builtin_nontemporal_store(val, &qk_out[qkbase + (long)m * Sq + n]);
      }
    }
  }
  __syncthreads();

  // ---- phase 2: masked softmax from LDS; attn_weights streamed once ----
#pragma unroll
  for (int rr = 0; rr < 2; ++rr) {               // each wave owns 2 rows
    int r = wave * 2 + rr;
    float mi = mrow[row0 + r];
    float mx = -3.4e38f;
    for (int j = lane; j < Sq; j += 32) {
      float mm = fminf(1.0f, mi + mrow[j]);
      mx = fmaxf(mx, P[r * Sq + j] + mm * (-1e9f));
    }
#pragma unroll
    for (int off = 16; off > 0; off >>= 1) mx = fmaxf(mx, __shfl_xor(mx, off, 32));
    float sum = 0.0f;
    for (int j = lane; j < Sq; j += 32) {
      float mm = fminf(1.0f, mi + mrow[j]);
      float e = __expf(P[r * Sq + j] + mm * (-1e9f) - mx);
      sum += e;
      W[r * Sq + j] = f2bf(e);                   // unnormalized probs for WMMA
    }
#pragma unroll
    for (int off = 16; off > 0; off >>= 1) sum += __shfl_xor(sum, off, 32);
    float inv = 1.0f / sum;
    if (lane == 0) rowinv[r] = inv;
    float* arow = attnw + qkbase + (long)r * Sq;
    for (int j = lane; j < Sq; j += 32) {        // exact f32 attn_weights
      float mm = fminf(1.0f, mi + mrow[j]);
      float e = __expf(P[r * Sq + j] + mm * (-1e9f) - mx);
      __builtin_nontemporal_store(e * inv, &arow[j]);
    }
  }
  __syncthreads();

  // ---- phase 3: P(16x2048) @ V(2048x64): 4 N-tiles x 2 K-halves / 8 waves ----
  int nt = wave & 3, kh = wave >> 2;
  const unsigned short* vrow =
      vT + ((long)(b * Hh + h) * HDm + nt * 16 + lr) * Sq;   // N-major, contiguous K
  v8f acc = {};
  for (int kk = kh * 1024; kk < kh * 1024 + 1024; kk += 32) {
    FragBf a, bf;
    int ao = lr * Sq + kk + hi * 8;
    a.q[0] = *(const u32x4*)&W[ao];
    a.q[1] = *(const u32x4*)&W[ao + 16];
    bf.q[0] = *(const u32x4*)(vrow + kk + hi * 8);
    bf.q[1] = *(const u32x4*)(vrow + kk + hi * 8 + 16);
    acc = __builtin_amdgcn_wmma_f32_16x16x32_bf16(false, a.v, false, bf.v,
                                                  (short)0, acc, false, false);
  }
  if (kh == 1) {
#pragma unroll
    for (int vg = 0; vg < 8; ++vg)
      partial[nt * 256 + (vg + hi * 8) * 16 + lr] = acc[vg];
  }
  __syncthreads();
  if (kh == 0) {
#pragma unroll
    for (int vg = 0; vg < 8; ++vg) {
      int m = vg + hi * 8;
      float val = (acc[vg] + partial[nt * 256 + m * 16 + lr]) * rowinv[m];
      long oi = ((long)b * Sq + row0 + m) * (Hh * HDm) + h * HDm + nt * 16 + lr;
      attnB[oi] = f2bf(val);                     // attn in (B,S,H*HD) bf16
    }
  }
}

// ---------------- launch ----------------
extern "C" void kernel_launch(void* const* d_in, const int* in_sizes, int n_in,
                              void* d_out, int out_size, void* d_ws, size_t ws_size,
                              hipStream_t stream) {
  const float* x     = (const float*)d_in[0];
  const float* mask  = (const float*)d_in[1];
  const float* wq_w  = (const float*)d_in[2];
  const float* wq_b  = (const float*)d_in[3];
  const float* wk_w  = (const float*)d_in[4];
  const float* wvc_w = (const float*)d_in[5];
  const float* wvr_w = (const float*)d_in[6];
  const float* wo_w  = (const float*)d_in[7];
  const float* wo_b  = (const float*)d_in[8];
  (void)in_sizes; (void)n_in; (void)out_size; (void)ws_size;

  float* out   = (float*)d_out + OUT_OFF;
  float* attnw = (float*)d_out + AW_OFF;
  float* qk    = (float*)d_out + QK_OFF;
  float* qout  = (float*)d_out + Q_OFF;
  float* kout  = (float*)d_out + K_OFF;
  float* vout  = (float*)d_out + V_OFF;

  char* ws = (char*)d_ws;
  size_t off = 0;
  auto take = [&](size_t bytes) -> char* {
    char* p = ws + off;
    off = (off + bytes + 255) & ~(size_t)255;
    return p;
  };
  unsigned short* xb    = (unsigned short*)take((size_t)Bsz * Sq * Dm * 2);
  unsigned short* wqT   = (unsigned short*)take((size_t)Dm * Hh * Lm * 2);
  unsigned short* wkT   = (unsigned short*)take((size_t)Dm * Lm * 2);
  unsigned short* wvcT  = (unsigned short*)take((size_t)Dm * Lm * 2);
  unsigned short* wvrT  = (unsigned short*)take((size_t)Lm * Hh * HDm * 2);
  unsigned short* woT   = (unsigned short*)take((size_t)Hh * HDm * Dm * 2);
  unsigned short* qb    = (unsigned short*)take((size_t)Bsz * Hh * Sq * Lm * 2);
  unsigned short* kb    = (unsigned short*)take((size_t)Bsz * Sq * Lm * 2);
  unsigned short* vcb   = (unsigned short*)take((size_t)Bsz * Sq * Lm * 2);
  unsigned short* vb    = (unsigned short*)take((size_t)Bsz * Hh * Sq * HDm * 2);
  unsigned short* vTb   = (unsigned short*)take((size_t)Bsz * Hh * HDm * Sq * 2);
  unsigned short* attb  = (unsigned short*)take((size_t)Bsz * Sq * Hh * HDm * 2);

  // x straight convert; weights transpose-convert to NxK for the NT GEMM
  {
    long n = (long)Bsz * Sq * Dm;
    conv_bf16_kernel<<<dim3((unsigned)((n + 255) / 256)), 256, 0, stream>>>(x, xb, n);
  }
  auto convT = [&](const float* s, unsigned short* d, int K, int N) {
    long n = (long)K * N;
    convT_bf16_kernel<<<dim3((unsigned)((n + 255) / 256)), 256, 0, stream>>>(s, d, K, N);
  };
  convT(wq_w,  wqT,  Dm, Hh * Lm);
  convT(wk_w,  wkT,  Dm, Lm);
  convT(wvc_w, wvcT, Dm, Lm);
  convT(wvr_w, wvrT, Lm, Hh * HDm);
  convT(wo_w,  woT,  Hh * HDm, Dm);

  const int Mx = Bsz * Sq; // 8192
  // q = x@wq + b  -> (B,H,S,L) f32 + bf16 copy
  gemm_bf16_wmma<<<dim3(Hh * Lm / 64, Mx / 64, 1), 256, 0, stream>>>(
      xb, Dm, wqT, Dm, qout, qb, wq_b, 1.0f, Mx, Hh * Lm, Dm, 1, Sq, Lm, Hh);
  // k = x@wk  -> (B,S,L) f32 + bf16
  gemm_bf16_wmma<<<dim3(Lm / 64, Mx / 64, 1), 256, 0, stream>>>(
      xb, Dm, wkT, Dm, kout, kb, nullptr, 1.0f, Mx, Lm, Dm, 0, 1, 1, 1);
  // vc = x@wvc -> bf16 only
  gemm_bf16_wmma<<<dim3(Lm / 64, Mx / 64, 1), 256, 0, stream>>>(
      xb, Dm, wvcT, Dm, nullptr, vcb, nullptr, 1.0f, Mx, Lm, Dm, 0, 1, 1, 1);
  // v = vc@wvr -> (B,H,S,HD) f32 + bf16
  gemm_bf16_wmma<<<dim3(Hh * HDm / 64, Mx / 64, 1), 256, 0, stream>>>(
      vcb, Lm, wvrT, Lm, vout, vb, nullptr, 1.0f, Mx, Hh * HDm, Lm, 1, Sq, HDm, Hh);
  // vT for N-major B fragments in the P@V WMMA
  {
    long tot = (long)Bsz * Hh * Sq * HDm;
    transpose_v_kernel<<<dim3((unsigned)((tot + 255) / 256)), 256, 0, stream>>>(vb, vTb);
  }
  // fused logits + mask + softmax + P@V
  {
    size_t smem = (size_t)16 * Sq * 4   // P  (f32)
                + (size_t)16 * Sq * 2   // W  (bf16)
                + (size_t)16 * Lm * 2   // Qs
                + 4 * 256 * 4           // partial
                + 16 * 4;               // rowinv  -> 204,864 B (< 320KB WGP LDS)
    mla_attn_fused<<<dim3(Sq / 16, Hh, Bsz), 256, smem, stream>>>(
        qb, kb, vTb, mask, qk, attnw, attb);
  }
  // out = attn@wo + b
  gemm_bf16_wmma<<<dim3(Dm / 64, Mx / 64, 1), 256, 0, stream>>>(
      attb, Hh * HDm, woT, Hh * HDm, out, nullptr, wo_b, 1.0f, Mx, Dm, Hh * HDm,
      0, 1, 1, 1);
}
